// Dense_16363825398113
// MI455X (gfx1250) — matrix operand order repaired and checked by reading, not compiled, not run on hardware
//
#include <hip/hip_runtime.h>

#define NDIM 512
#define BROWS 131072

typedef __attribute__((ext_vector_type(16))) __bf16 v16bf;
typedef __attribute__((ext_vector_type(8)))  float  v8f;

union Frag { v16bf v; uint4 q[2]; };

static __device__ __forceinline__ unsigned short f32_to_bf16_rne(float f) {
  unsigned int u = __float_as_uint(f);
  u += 0x7FFFu + ((u >> 16) & 1u);
  return (unsigned short)(u >> 16);
}
static __device__ __forceinline__ float bf16_to_f32(unsigned short h) {
  return __uint_as_float(((unsigned int)h) << 16);
}
static __device__ __forceinline__ unsigned int pack_bf16x2(float a, float b) {
  return (unsigned int)f32_to_bf16_rne(a) | ((unsigned int)f32_to_bf16_rne(b) << 16);
}

// ---------- expm via degree-13 Taylor (Horner) ----------

// T = I + A/13
__global__ void init_T13(const float* __restrict__ A, float* __restrict__ T) {
  int idx = blockIdx.x * blockDim.x + threadIdx.x;
  int r = idx >> 9, c = idx & (NDIM - 1);
  T[idx] = A[idx] * (1.0f / 13.0f) + (r == c ? 1.0f : 0.0f);
}

// out = I + (A * T) * invk   (512x512 f32, LDS tiled)
__global__ void horner_mm(const float* __restrict__ A, const float* __restrict__ T,
                          float* __restrict__ out, float invk) {
  __shared__ float As[16][17];
  __shared__ float Ts[16][17];
  const int tx = threadIdx.x, ty = threadIdx.y;
  const int row = blockIdx.y * 16 + ty;
  const int col = blockIdx.x * 16 + tx;
  float acc = 0.0f;
  for (int kt = 0; kt < NDIM; kt += 16) {
    As[ty][tx] = A[row * NDIM + kt + tx];
    Ts[ty][tx] = T[(kt + ty) * NDIM + col];
    __syncthreads();
#pragma unroll
    for (int k = 0; k < 16; ++k) acc += As[ty][k] * Ts[k][tx];
    __syncthreads();
  }
  out[row * NDIM + col] = acc * invk + (row == col ? 1.0f : 0.0f);
}

// E -> (E_hi, E_lo) bf16 split
__global__ void split_bf16(const float* __restrict__ E,
                           unsigned short* __restrict__ hi,
                           unsigned short* __restrict__ lo) {
  int idx = blockIdx.x * blockDim.x + threadIdx.x;
  float v = E[idx];
  unsigned short h = f32_to_bf16_rne(v);
  hi[idx] = h;
  lo[idx] = f32_to_bf16_rne(v - bf16_to_f32(h));
}

// loss = -trace(A)/N
__global__ void trace_loss(const float* __restrict__ A, float* __restrict__ outLoss) {
  __shared__ float s[256];
  int t = threadIdx.x;
  s[t] = A[t * (NDIM + 1)] + A[(t + 256) * (NDIM + 1)];
  __syncthreads();
  for (int off = 128; off > 0; off >>= 1) {
    if (t < off) s[t] += s[t + off];
    __syncthreads();
  }
  if (t == 0) outLoss[0] = -s[0] / (float)NDIM;
}

// ---------- big GEMM: out = (x + bias) @ E^T via bf16x3 WMMA ----------
#define WMMA_BF16(ACC, A_, B_) \
  ACC = __builtin_amdgcn_wmma_f32_16x16x32_bf16(false, (A_).v, false, (B_).v, (short)0, ACC, false, false)

__global__ __launch_bounds__(32) void gemm_bf16x3(
    const float* __restrict__ x, const float* __restrict__ bias,
    const unsigned short* __restrict__ Ehi, const unsigned short* __restrict__ Elo,
    float* __restrict__ out) {
  __shared__ unsigned short xh[32 * NDIM];  // 32 KB
  __shared__ unsigned short xl[32 * NDIM];  // 32 KB

  const int lane = threadIdx.x;             // 0..31 (wave32)
  const size_t r0 = (size_t)blockIdx.x * 32;

  // Stage 32 rows of (x + bias) into LDS as bf16 hi/lo (4 elems / lane / iter).
#pragma unroll 4
  for (int i = 0; i < 128; ++i) {
    int t = i * 32 + lane;                  // quad index 0..4095
    int k4 = (t & 127) * 4;                 // 0,4,...,508
    int row = t >> 7;                       // 0..31
    const float4 xv = *(const float4*)(x + (r0 + row) * NDIM + k4);
    const float4 bv = *(const float4*)(bias + k4);
    float v0 = xv.x + bv.x, v1 = xv.y + bv.y, v2 = xv.z + bv.z, v3 = xv.w + bv.w;
    float h0 = bf16_to_f32(f32_to_bf16_rne(v0));
    float h1 = bf16_to_f32(f32_to_bf16_rne(v1));
    float h2 = bf16_to_f32(f32_to_bf16_rne(v2));
    float h3 = bf16_to_f32(f32_to_bf16_rne(v3));
    uint2 ph, pl;
    ph.x = pack_bf16x2(v0, v1);
    ph.y = pack_bf16x2(v2, v3);
    pl.x = pack_bf16x2(v0 - h0, v1 - h1);
    pl.y = pack_bf16x2(v2 - h2, v3 - h3);
    *(uint2*)(&xh[row * NDIM + k4]) = ph;
    *(uint2*)(&xl[row * NDIM + k4]) = pl;
  }
  __syncthreads();

  const int m = lane & 15;                  // A row within tile
  const int koffA = (lane >> 4) * 16;       // A K-offset per lane half
  const int n16 = lane & 15;                // B column within tile
  const int koffB = (lane >> 4) * 16;       // B K-offset per lane half

  // 2 row-tiles x 2 n-tiles per iteration -> 4 independent accumulator chains.
#pragma unroll 1
  for (int nt = 0; nt < 32; nt += 2) {
    const unsigned short* eh0 = Ehi + (size_t)(nt * 16 + n16) * NDIM + koffB;
    const unsigned short* el0 = Elo + (size_t)(nt * 16 + n16) * NDIM + koffB;
    const unsigned short* eh1 = Ehi + (size_t)((nt + 1) * 16 + n16) * NDIM + koffB;
    const unsigned short* el1 = Elo + (size_t)((nt + 1) * 16 + n16) * NDIM + koffB;
    v8f acc00 = {};  // rows 0-15,  cols nt
    v8f acc10 = {};  // rows 16-31, cols nt
    v8f acc01 = {};  // rows 0-15,  cols nt+1
    v8f acc11 = {};  // rows 16-31, cols nt+1
#pragma unroll
    for (int ks = 0; ks < NDIM; ks += 32) {
      Frag ah0, al0, ah1, al1, bh0, bl0, bh1, bl1;
      const uint4* pa0h = (const uint4*)(&xh[m * NDIM + ks + koffA]);
      const uint4* pa0l = (const uint4*)(&xl[m * NDIM + ks + koffA]);
      const uint4* pa1h = (const uint4*)(&xh[(m + 16) * NDIM + ks + koffA]);
      const uint4* pa1l = (const uint4*)(&xl[(m + 16) * NDIM + ks + koffA]);
      ah0.q[0] = pa0h[0]; ah0.q[1] = pa0h[1];   // +8 elements = +16B
      al0.q[0] = pa0l[0]; al0.q[1] = pa0l[1];
      ah1.q[0] = pa1h[0]; ah1.q[1] = pa1h[1];
      al1.q[0] = pa1l[0]; al1.q[1] = pa1l[1];
      const uint4* pbh0 = (const uint4*)(eh0 + ks);
      const uint4* pbl0 = (const uint4*)(el0 + ks);
      const uint4* pbh1 = (const uint4*)(eh1 + ks);
      const uint4* pbl1 = (const uint4*)(el1 + ks);
      bh0.q[0] = pbh0[0]; bh0.q[1] = pbh0[1];
      bl0.q[0] = pbl0[0]; bl0.q[1] = pbl0[1];
      bh1.q[0] = pbh1[0]; bh1.q[1] = pbh1[1];
      bl1.q[0] = pbl1[0]; bl1.q[1] = pbl1[1];

      // hi*hi terms (4 independent chains)
      WMMA_BF16(acc00, ah0, bh0);
      WMMA_BF16(acc10, ah1, bh0);
      WMMA_BF16(acc01, ah0, bh1);
      WMMA_BF16(acc11, ah1, bh1);
      // hi*lo terms
      WMMA_BF16(acc00, ah0, bl0);
      WMMA_BF16(acc10, ah1, bl0);
      WMMA_BF16(acc01, ah0, bl1);
      WMMA_BF16(acc11, ah1, bl1);
      // lo*hi terms
      WMMA_BF16(acc00, al0, bh0);
      WMMA_BF16(acc10, al1, bh0);
      WMMA_BF16(acc01, al0, bh1);
      WMMA_BF16(acc11, al1, bh1);
    }
    // C/D layout: VGPR i -> row ((lane>=16?8:0)+i), col = lane&15
    const int rr = (lane >> 4) * 8;
#pragma unroll
    for (int i = 0; i < 8; ++i) {
      out[(r0 + rr + i) * NDIM + nt * 16 + n16] = acc00[i];
      out[(r0 + 16 + rr + i) * NDIM + nt * 16 + n16] = acc10[i];
      out[(r0 + rr + i) * NDIM + (nt + 1) * 16 + n16] = acc01[i];
      out[(r0 + 16 + rr + i) * NDIM + (nt + 1) * 16 + n16] = acc11[i];
    }
  }
}

extern "C" void kernel_launch(void* const* d_in, const int* in_sizes, int n_in,
                              void* d_out, int out_size, void* d_ws, size_t ws_size,
                              hipStream_t stream) {
  const float* x = (const float*)d_in[0];
  const float* A = (const float*)d_in[1];       // matrix [512,512]
  const float* bias = (const float*)d_in[2];
  float* out = (float*)d_out;                   // [B*N] out, then 1 loss float

  char* w = (char*)d_ws;
  float* P0 = (float*)w;                                          // 1 MB
  float* P1 = (float*)(w + (size_t)NDIM * NDIM * 4);              // 1 MB
  unsigned short* Ehi = (unsigned short*)(w + (size_t)NDIM * NDIM * 8);                           // 0.5 MB
  unsigned short* Elo = (unsigned short*)(w + (size_t)NDIM * NDIM * 8 + (size_t)NDIM * NDIM * 2); // 0.5 MB

  // E = expm(A) via degree-13 Taylor, Horner form: T = I + (A*T)/k
  init_T13<<<(NDIM * NDIM) / 256, 256, 0, stream>>>(A, P0);
  float* src = P0;
  float* dst = P1;
  for (int k = 12; k >= 1; --k) {
    horner_mm<<<dim3(32, 32), dim3(16, 16), 0, stream>>>(A, src, dst, 1.0f / (float)k);
    float* tmp = src; src = dst; dst = tmp;
  }
  // final E in src (12 swaps -> back in P0)
  split_bf16<<<(NDIM * NDIM) / 256, 256, 0, stream>>>(src, Ehi, Elo);
  trace_loss<<<1, 256, 0, stream>>>(A, out + (size_t)BROWS * NDIM);
  gemm_bf16x3<<<BROWS / 32, 32, 0, stream>>>(x, bias, Ehi, Elo, out);
}
